// YemongSpatial_807453851686
// MI455X (gfx1250) — matrix operand order repaired and checked by reading, not compiled, not run on hardware
//
#include <hip/hip_runtime.h>

#define DEV __device__ __forceinline__

typedef _Float16 h16v __attribute__((ext_vector_type(16)));
typedef _Float16 h8v  __attribute__((ext_vector_type(8)));
typedef float    f8v  __attribute__((ext_vector_type(8)));

// Problem sizes
constexpr int Bc = 32, Tc = 512, Nc = 8;
constexpr int DS = 16, DM = 256, NH = 8, NL = 4, DR = 64, AD = 12;
constexpr int NTOK = Bc * Tc * Nc;           // 131072 tokens

// Tiling
constexpr int GPW   = 4;                     // (b,t) groups per workgroup
constexpr int MROWS = GPW * Nc;              // 32 token rows per WG
constexpr int NXS   = DM + 8;                // f16 LDS row stride (bank-conflict-free A-frags)
constexpr int XS    = DM + 4;                // f32 LDS row stride

// d_ws layout (f16 elements): [encW2T][ per layer: WqT WkT WvT WoT ][headWT 16x256]
constexpr int WSQ      = DM * DM;            // 65536
constexpr int OFF_HEAD = 17 * WSQ;
constexpr int WS_ELEMS = 17 * WSQ + 16 * DM;

// ---------------------------------------------------------------------------
// Prep: fp32 weights -> f16, transposed to out-major [o][k] for WMMA B-frags
// ---------------------------------------------------------------------------
__global__ void prep_weights(const float* __restrict__ encW2,
                             const float* __restrict__ Wq,
                             const float* __restrict__ Wk,
                             const float* __restrict__ Wv,
                             const float* __restrict__ Wo,
                             const float* __restrict__ headW,
                             _Float16* __restrict__ ws) {
  int e = blockIdx.x * blockDim.x + threadIdx.x;
  if (e < OFF_HEAD) {
    int mat = e / WSQ;
    int o   = (e % WSQ) / DM;      // output column (row of transposed)
    int k   = e % DM;
    const float* src;
    if (mat == 0) {
      src = encW2;
    } else {
      int m = mat - 1, l = m >> 2, w = m & 3;
      const float* base = (w == 0) ? Wq : (w == 1) ? Wk : (w == 2) ? Wv : Wo;
      src = base + (size_t)l * WSQ;
    }
    ws[e] = (_Float16)src[k * DM + o];
  } else if (e < WS_ELEMS) {
    int e2 = e - OFF_HEAD;
    int o = e2 / DM, k = e2 % DM;
    ws[e] = (_Float16)((o < AD) ? headW[k * AD + o] : 0.0f);
  }
}

// ---------------------------------------------------------------------------
// Fragment loads: 16-byte vector loads concatenated in registers (b128 paths)
// ---------------------------------------------------------------------------
DEV h16v frag_cat(h8v lo, h8v hi) {
  return __builtin_shufflevector(lo, hi, 0, 1, 2, 3, 4, 5, 6, 7,
                                 8, 9, 10, 11, 12, 13, 14, 15);
}

// A (LDS, 16-bit 16x32 map): lane -> row M=lane%16; K runs [kb..kb+7],[kb+16..kb+23]
DEV h16v load_a_frag(const _Float16* p) {
  h8v lo = *(const h8v*)p;
  h8v hi = *(const h8v*)(p + 16);
  return frag_cat(lo, hi);
}

// B (global, out-major WT): lane -> col N=lane%16; K runs [kb..kb+7],[kb+16..kb+23], kb in {0,8}
DEV h16v load_b_frag(const _Float16* __restrict__ p) {
  h8v lo = *(const h8v*)p;
  h8v hi = *(const h8v*)(p + 16);
  return frag_cat(lo, hi);
}

// WMMA tile engine: acc[NT] (16x16 f32) += X(LDS f16) @ WT(global f16)
template <int NT>
DEV void gemm_tiles(const _Float16* Xlds, const _Float16* __restrict__ WT,
                    f8v* acc, int mt, int ntb, int lane) {
  const int m   = mt * 16 + (lane & 15);
  const _Float16* arow = Xlds + m * NXS + ((lane < 16) ? 0 : 8);
  const int nb  = (lane & 15);
  const int kbB = (lane < 16) ? 0 : 8;
  const _Float16* brow = WT + nb * DM + kbB;
#pragma unroll
  for (int ks = 0; ks < 8; ++ks) {
    const int k0 = ks * 32;
    h16v a = load_a_frag(arow + k0);
#pragma unroll
    for (int t = 0; t < NT; ++t) {
      h16v b = load_b_frag(brow + (ntb + t) * 16 * DM + k0);
      acc[t] = __builtin_amdgcn_wmma_f32_16x16x32_f16(
          false, a, false, b, (short)0, acc[t], false, false);
    }
  }
}

// ---------------------------------------------------------------------------
// Fused forward: encoder + trunk + 4 attention layers + head. One WG = 4 (b,t)
// groups (32 tokens); everything stays in LDS.
// ---------------------------------------------------------------------------
__global__ __launch_bounds__(256) void fwd_kernel(
    const float* __restrict__ state, const float* __restrict__ pos,
    const float* __restrict__ vel, const int* __restrict__ team,
    const unsigned char* __restrict__ alive, const float* __restrict__ dead,
    const float* __restrict__ encW1, const float* __restrict__ encb1,
    const float* __restrict__ encb2, const float* __restrict__ team_emb,
    const float* __restrict__ ship_emb, const float* __restrict__ trunkW1,
    const float* __restrict__ trunkb1, const float* __restrict__ trunkW2,
    const float* __restrict__ trunkb2, const float* __restrict__ adaptW,
    const float* __restrict__ adaptb, const float* __restrict__ normw,
    const float* __restrict__ bq, const float* __restrict__ bk,
    const float* __restrict__ bv, const float* __restrict__ bo,
    const float* __restrict__ headb, const _Float16* __restrict__ ws,
    float* __restrict__ out) {
  __shared__ alignas(16) float     xb[MROWS * XS];             // residual x (f32)
  __shared__ alignas(16) _Float16  nxb[MROWS * NXS];           // norm-x / attn-out
  __shared__ alignas(16) _Float16  qb[MROWS * NXS];
  __shared__ alignas(16) _Float16  kbuf[MROWS * NXS];
  __shared__ alignas(16) _Float16  vbuf[MROWS * NXS];          // also trunkW2 stage
  __shared__ alignas(16) _Float16  trunkS[GPW * Nc * Nc * DR]; // pairwise trunk
  __shared__ alignas(16) float     sst[MROWS][DS];
  __shared__ alignas(16) float     adWs[DR * NH];              // adapt_W[l] stage
  __shared__ alignas(16) float     posL[MROWS][2], velL[MROWS][2];
  __shared__ int       aliveL[MROWS], teamL[MROWS];
  __shared__ alignas(16) float     red[MROWS][8];

  const int tid  = threadIdx.x;
  const int lane = tid & 31;
  const int wave = tid >> 5;
  const int tok0 = blockIdx.x * MROWS;
  const int mt   = wave & 1;
  const int ntb  = (wave >> 1) * 4;

  // ---- stage per-token metadata ----
  if (tid < MROWS) {
    int tok = tok0 + tid;
    aliveL[tid] = alive[tok];
    teamL[tid]  = team[tok];
    posL[tid][0] = pos[tok * 2 + 0]; posL[tid][1] = pos[tok * 2 + 1];
    velL[tid][0] = vel[tok * 2 + 0]; velL[tid][1] = vel[tok * 2 + 1];
  }
  __syncthreads();
  // masked state
  for (int e = tid; e < MROWS * DS; e += 256) {
    int r = e >> 4, d = e & 15;
    sst[r][d] = aliveL[r] ? state[(tok0 + r) * DS + d] : dead[d];
  }
  // prefetch layer-0 qkv weights (f16) into cache hierarchy
  {
    const char* p0 = (const char*)(ws + WSQ);
    for (int i = tid; i < (3 * WSQ * 2) / 128; i += 256)
      __builtin_prefetch(p0 + i * 128, 0, 1);
  }
  __syncthreads();

  // ---- encoder MLP layer 1 (K=16, VALU) -> nxb = silu(state@W1+b1) f16 ----
  {
    const int r = tid >> 3, c0 = (tid & 7) * 32;
    float srow[DS];
#pragma unroll
    for (int d = 0; d < DS; ++d) srow[d] = sst[r][d];
    for (int c = c0; c < c0 + 32; ++c) {
      float z = encb1[c];
#pragma unroll
      for (int d = 0; d < DS; ++d) z += srow[d] * encW1[d * DM + c];
      nxb[r * NXS + c] = (_Float16)(z / (1.0f + __expf(-z)));
    }
  }
  __syncthreads();

  // ---- encoder layer 2 via WMMA: x = nxb @ encW2 + b2 + team/ship embeds ----
  {
    f8v acc[4];
#pragma unroll
    for (int t = 0; t < 4; ++t)
#pragma unroll
      for (int r = 0; r < 8; ++r) acc[t][r] = 0.0f;
    gemm_tiles<4>(nxb, ws, acc, mt, ntb, lane);
    const int nb = lane & 15, mb = mt * 16 + ((lane < 16) ? 0 : 8);
#pragma unroll
    for (int t = 0; t < 4; ++t) {
      const int n = (ntb + t) * 16 + nb;
      const float b2v = encb2[n];
#pragma unroll
      for (int r = 0; r < 8; ++r) {
        const int m = mb + r;
        xb[m * XS + n] = acc[t][r] + b2v + team_emb[teamL[m] * DM + n] +
                         ship_emb[(m & 7) * DM + n];
      }
    }
  }
  // stage trunkW2 (64x64 f32 = 16KB) into vbuf region (free until QKV GEMMs)
  {
    float* w2s = (float*)vbuf;
    for (int e = tid; e < DR * DR; e += 256) w2s[e] = trunkW2[e];
  }
  __syncthreads();

  // ---- pairwise trunk MLP (one pair per thread, register-resident t1) ----
  {
    const float* w2s = (const float*)vbuf;
    const int g = tid >> 6, i = (tid >> 3) & 7, j = tid & 7;
    const int ri = g * 8 + i, rj = g * 8 + j;
    const float W0 = 1024.0f;
    float px = (posL[ri][0] - posL[rj][0]) * W0;
    float py = (posL[ri][1] - posL[rj][1]) * W0;
    px -= rintf(px / W0) * W0;
    py -= rintf(py / W0) * W0;
    const float vx = velL[ri][0] - velL[rj][0];
    const float vy = velL[ri][1] - velL[rj][1];
    const float dn = sqrtf(px * px + py * py + 1e-8f) / W0;
    float f[6] = {px / W0, py / W0, vx, vy, dn, 1.0f / (1.0f + dn)};
    float t1[DR];
#pragma unroll
    for (int k = 0; k < DR; ++k) {
      float z = trunkb1[k];
#pragma unroll
      for (int q = 0; q < 6; ++q) z += f[q] * trunkW1[q * DR + k];
      t1[k] = z / (1.0f + __expf(-z));
    }
    _Float16* td = trunkS + ((g * Nc + i) * Nc + j) * DR;
    for (int k = 0; k < DR; ++k) {
      float z = trunkb2[k];
#pragma unroll
      for (int q = 0; q < DR; ++q) z += t1[q] * w2s[q * DR + k];  // LDS broadcast
      td[k] = (_Float16)z;
    }
  }
  __syncthreads();

  // ---- transformer layers ----
  for (int l = 0; l < NL; ++l) {
    // RMS partial sums + stage adapt_W[l] into LDS
    {
      const int r = tid >> 3, sg = tid & 7, c0 = sg * 32;
      float s = 0.0f;
      for (int c = c0; c < c0 + 32; ++c) { float v = xb[r * XS + c]; s += v * v; }
      red[r][sg] = s;
      for (int e = tid; e < DR * NH; e += 256) adWs[e] = adaptW[l * DR * NH + e];
    }
    __syncthreads();
    {
      const int r = tid >> 3, sg = tid & 7, c0 = sg * 32;
      float s = 0.0f;
#pragma unroll
      for (int p = 0; p < 8; ++p) s += red[r][p];
      const float inv = rsqrtf(s * (1.0f / DM) + 1e-6f);
      for (int c = c0; c < c0 + 32; ++c)
        nxb[r * NXS + c] = (_Float16)(xb[r * XS + c] * inv * normw[l * DM + c]);
    }
    __syncthreads();

    // Q, K, V projections via WMMA
    {
      const int nb = lane & 15, mb = mt * 16 + ((lane < 16) ? 0 : 8);
      _Float16* dsts[3] = {qb, kbuf, vbuf};
      const float* bss[3] = {bq + l * DM, bk + l * DM, bv + l * DM};
#pragma unroll
      for (int p = 0; p < 3; ++p) {
        const _Float16* WT = ws + (size_t)WSQ * (1 + l * 4 + p);
        f8v acc[4];
#pragma unroll
        for (int t = 0; t < 4; ++t)
#pragma unroll
          for (int r = 0; r < 8; ++r) acc[t][r] = 0.0f;
        gemm_tiles<4>(nxb, WT, acc, mt, ntb, lane);
#pragma unroll
        for (int t = 0; t < 4; ++t) {
          const int n = (ntb + t) * 16 + nb;
          const float bvv = bss[p][n];
#pragma unroll
          for (int r = 0; r < 8; ++r)
            dsts[p][(mb + r) * NXS + n] = (_Float16)(acc[t][r] + bvv);
        }
      }
    }
    __syncthreads();

    // Attention: one (group, head, query-row) per thread; 8x8 scores + adapt bias
    {
      const int g = tid >> 6, h = (tid >> 3) & 7, i = tid & 7;
      const int ri = g * 8 + i;
      const float scale = 0.17677669529663687f;  // 32^-0.5
      float qf[32];
      {
        const _Float16* qrow = qb + ri * NXS + h * 32;
#pragma unroll
        for (int c = 0; c < 4; ++c) {
          h8v qv = *(const h8v*)(qrow + c * 8);
#pragma unroll
          for (int e = 0; e < 8; ++e) qf[c * 8 + e] = (float)qv[e];
        }
      }
      const float ab = adaptb[l * NH + h];
      float sj[8];
      float mx = -3.402823e38f;
#pragma unroll
      for (int j = 0; j < 8; ++j) {
        const int rj = g * 8 + j;
        const _Float16* krow = kbuf + rj * NXS + h * 32;
        float d0 = 0.0f;
#pragma unroll
        for (int c = 0; c < 4; ++c) {
          h8v kv = *(const h8v*)(krow + c * 8);
#pragma unroll
          for (int e = 0; e < 8; ++e) d0 += qf[c * 8 + e] * (float)kv[e];
        }
        const _Float16* tr = trunkS + ((g * Nc + i) * Nc + j) * DR;
        float bia = ab;
        for (int k = 0; k < DR; ++k) bia += (float)tr[k] * adWs[k * NH + h];
        const float sc = aliveL[rj] ? (d0 * scale + bia) : -1.0e9f;
        sj[j] = sc;
        mx = fmaxf(mx, sc);
      }
      float se = 0.0f;
#pragma unroll
      for (int j = 0; j < 8; ++j) { sj[j] = __expf(sj[j] - mx); se += sj[j]; }
      const float isum = 1.0f / se;
      float o[32];
#pragma unroll
      for (int d = 0; d < 32; ++d) o[d] = 0.0f;
#pragma unroll
      for (int j = 0; j < 8; ++j) {
        const float p = sj[j] * isum;
        const _Float16* vrow = vbuf + (g * 8 + j) * NXS + h * 32;
#pragma unroll
        for (int c = 0; c < 4; ++c) {
          h8v vv = *(const h8v*)(vrow + c * 8);
#pragma unroll
          for (int e = 0; e < 8; ++e) o[c * 8 + e] += p * (float)vv[e];
        }
      }
      _Float16* orow = nxb + ri * NXS + h * 32;  // reuse nxb as attn-out
#pragma unroll
      for (int c = 0; c < 4; ++c) {
        h8v ov;
#pragma unroll
        for (int e = 0; e < 8; ++e) ov[e] = (_Float16)o[c * 8 + e];
        *(h8v*)(orow + c * 8) = ov;
      }
    }
    __syncthreads();

    // Output projection via WMMA, accumulate residual into xb
    {
      const _Float16* WoT = ws + (size_t)WSQ * (1 + l * 4 + 3);
      f8v acc[4];
#pragma unroll
      for (int t = 0; t < 4; ++t)
#pragma unroll
        for (int r = 0; r < 8; ++r) acc[t][r] = 0.0f;
      gemm_tiles<4>(nxb, WoT, acc, mt, ntb, lane);
      const int nb = lane & 15, mb = mt * 16 + ((lane < 16) ? 0 : 8);
#pragma unroll
      for (int t = 0; t < 4; ++t) {
        const int n = (ntb + t) * 16 + nb;
        const float bvv = bo[l * DM + n];
#pragma unroll
        for (int r = 0; r < 8; ++r) xb[(mb + r) * XS + n] += acc[t][r] + bvv;
      }
    }
    __syncthreads();
  }

  // ---- head: logits = x @ head_W + head_b (WMMA, padded 16-col tile) ----
  {
    const int r = tid >> 3, sg = tid & 7, c0 = sg * 32;
    for (int c = c0; c < c0 + 32; ++c)
      nxb[r * NXS + c] = (_Float16)xb[r * XS + c];
  }
  __syncthreads();
  if (wave < 2) {
    f8v acc;
#pragma unroll
    for (int r = 0; r < 8; ++r) acc[r] = 0.0f;
    gemm_tiles<1>(nxb, ws + OFF_HEAD, &acc, wave, 0, lane);
    const int nb = lane & 15, mb = wave * 16 + ((lane < 16) ? 0 : 8);
    if (nb < AD) {
      const float hb = headb[nb];
#pragma unroll
      for (int r = 0; r < 8; ++r)
        out[(size_t)(tok0 + mb + r) * AD + nb] = acc[r] + hb;
    }
  }
}

// ---------------------------------------------------------------------------
extern "C" void kernel_launch(void* const* d_in, const int* in_sizes, int n_in,
                              void* d_out, int out_size, void* d_ws,
                              size_t ws_size, hipStream_t stream) {
  const float* state   = (const float*)d_in[0];
  const float* pos     = (const float*)d_in[1];
  const float* vel     = (const float*)d_in[2];
  const int*   team    = (const int*)d_in[3];
  const unsigned char* alive = (const unsigned char*)d_in[4];
  const float* dead    = (const float*)d_in[5];
  const float* encW1   = (const float*)d_in[6];
  const float* encb1   = (const float*)d_in[7];
  const float* encW2   = (const float*)d_in[8];
  const float* encb2   = (const float*)d_in[9];
  const float* team_emb = (const float*)d_in[10];
  const float* ship_emb = (const float*)d_in[11];
  const float* trunkW1 = (const float*)d_in[12];
  const float* trunkb1 = (const float*)d_in[13];
  const float* trunkW2 = (const float*)d_in[14];
  const float* trunkb2 = (const float*)d_in[15];
  const float* adaptW  = (const float*)d_in[16];
  const float* adaptb  = (const float*)d_in[17];
  const float* normw   = (const float*)d_in[18];
  const float* Wq      = (const float*)d_in[19];
  const float* bq      = (const float*)d_in[20];
  const float* Wk      = (const float*)d_in[21];
  const float* bk      = (const float*)d_in[22];
  const float* Wv      = (const float*)d_in[23];
  const float* bv      = (const float*)d_in[24];
  const float* Wo      = (const float*)d_in[25];
  const float* bo      = (const float*)d_in[26];
  const float* headW   = (const float*)d_in[27];
  const float* headb   = (const float*)d_in[28];
  _Float16* ws = (_Float16*)d_ws;
  float* out = (float*)d_out;

  prep_weights<<<(WS_ELEMS + 255) / 256, 256, 0, stream>>>(
      encW2, Wq, Wk, Wv, Wo, headW, ws);

  const int nblocks = (NTOK / Nc) / GPW;  // 16384 / 4 = 4096
  fwd_kernel<<<nblocks, 256, 0, stream>>>(
      state, pos, vel, team, alive, dead, encW1, encb1, encb2, team_emb,
      ship_emb, trunkW1, trunkb1, trunkW2, trunkb2, adaptW, adaptb, normw,
      bq, bk, bv, bo, headb, ws, out);
}
